// SelfAttentionBlock_73340861547104
// MI455X (gfx1250) — matrix operand order repaired and checked by reading, not compiled, not run on hardware
//
#include <hip/hip_runtime.h>
#include <math.h>

// ---------------------------------------------------------------------------
// Self-attention block for MI455X (gfx1250).
// bf16 WMMA everywhere, f32 accumulate. One bf16 conversion pre-pass so the
// GEMM inner loops are pure global_load_b128 / ds_load_b128 + v_wmma.
// B=8, C=512, T=1024, 8 heads x 64.
// ---------------------------------------------------------------------------

typedef __attribute__((ext_vector_type(16))) __bf16 v16bf;
typedef __attribute__((ext_vector_type(8)))  __bf16 v8bf;
typedef __attribute__((ext_vector_type(8)))  float  v8f;

constexpr int kB  = 8;
constexpr int kC  = 512;
constexpr int kT  = 1024;           // H*W
constexpr int kNH = 8;
constexpr int kHD = kC / kNH;       // 64

// --- WMMA operand loaders -------------------------------------------------
// A-matrix (16x32 bf16), ISA 7.12.2: lane half hi needs k = 8hi..8hi+7 and
// 16+8hi..16+8hi+7 of its row -> two 16B loads at base(+0) and base(+16),
// where base = &row[k0 + 8*hi].
__device__ __forceinline__ v16bf ldA(const __bf16* base) {
  v8bf lo = *(const v8bf*)(base);
  v8bf hi = *(const v8bf*)(base + 16);
  return __builtin_shufflevector(lo, hi, 0,1,2,3,4,5,6,7,8,9,10,11,12,13,14,15);
}
// B-matrix (32x16 bf16): lane half hi needs k = 8hi..8hi+7 and
// 16+8hi..16+8hi+7 of its column -> two 16B loads at base(+0) and base(+16),
// base = &col[k0 + 8*hi].
__device__ __forceinline__ v16bf ldB(const __bf16* base) {
  v8bf lo = *(const v8bf*)(base);
  v8bf hi = *(const v8bf*)(base + 16);
  return __builtin_shufflevector(lo, hi, 0,1,2,3,4,5,6,7,8,9,10,11,12,13,14,15);
}

__device__ __forceinline__ v8f wmma_bf16(v16bf a, v16bf b, v8f c) {
  return __builtin_amdgcn_wmma_f32_16x16x32_bf16(false, a, false, b, (short)0, c,
                                                 false, false);
}

// ---------------------------------------------------------------------------
// Pre-pass A: transpose+convert f32 (nrow x ncol) -> bf16 (ncol x nrow),
// batched over blockIdx.z. 32x32 LDS tile, 256 threads.
// Used for x: (C,T) -> xbf (T,C).
// ---------------------------------------------------------------------------
__global__ void __launch_bounds__(256)
xpose_x_kernel(const float* __restrict__ src, __bf16* __restrict__ dst)
{
  __shared__ float tile[32][33];
  const int nrow = kC, ncol = kT;
  const int ct = blockIdx.x * 32;            // col tile (in src)
  const int rt = blockIdx.y * 32;            // row tile (in src)
  const float* s = src + (size_t)blockIdx.z * nrow * ncol;
  __bf16* d      = dst + (size_t)blockIdx.z * nrow * ncol;
  const int tx = threadIdx.x & 31, ty = threadIdx.x >> 5;
#pragma unroll
  for (int rr = ty; rr < 32; rr += 8)
    tile[rr][tx] = s[(size_t)(rt + rr) * ncol + ct + tx];
  __syncthreads();
#pragma unroll
  for (int rr = ty; rr < 32; rr += 8)
    d[(size_t)(ct + rr) * nrow + rt + tx] = (__bf16)tile[tx][rr];
}

// Pre-pass B: the four C x C weights, transposed to (Cout, Cin) bf16.
__global__ void __launch_bounds__(256)
xpose_w_kernel(const float* __restrict__ Wq, const float* __restrict__ Wk,
               const float* __restrict__ Wv, const float* __restrict__ Wo,
               __bf16* __restrict__ Wqt, __bf16* __restrict__ Wkt,
               __bf16* __restrict__ Wvt, __bf16* __restrict__ Wot)
{
  __shared__ float tile[32][33];
  const int sel = blockIdx.z;
  const float* s = (sel == 0) ? Wq : (sel == 1) ? Wk : (sel == 2) ? Wv : Wo;
  __bf16* d      = (sel == 0) ? Wqt : (sel == 1) ? Wkt : (sel == 2) ? Wvt : Wot;
  const int ct = blockIdx.x * 32;
  const int rt = blockIdx.y * 32;
  const int tx = threadIdx.x & 31, ty = threadIdx.x >> 5;
#pragma unroll
  for (int rr = ty; rr < 32; rr += 8)
    tile[rr][tx] = s[(size_t)(rt + rr) * kC + ct + tx];
  __syncthreads();
#pragma unroll
  for (int rr = ty; rr < 32; rr += 8)
    d[(size_t)(ct + rr) * kC + rt + tx] = (__bf16)tile[tx][rr];
}

// ---------------------------------------------------------------------------
// Kernel 1: fused QKV projection (all-bf16 operands).
// grid = (T/16, C/64, B*3), block = 32 (one wave); 16(T) x 64(Cout) per wave.
// Q,K stored (B,NH,T,HD); V stored transposed (B,NH,HD,T).
// ---------------------------------------------------------------------------
__global__ void __launch_bounds__(32)
qkv_proj_kernel(const __bf16* __restrict__ xbf,
                const __bf16* __restrict__ Wqt, const float* __restrict__ bq,
                const __bf16* __restrict__ Wkt, const float* __restrict__ bk,
                const __bf16* __restrict__ Wvt, const float* __restrict__ bv,
                __bf16* __restrict__ Qb, __bf16* __restrict__ Kb,
                __bf16* __restrict__ Vt)
{
  const int lane = threadIdx.x & 31;
  const int r  = lane & 15;
  const int hi = lane >> 4;
  const int m0 = blockIdx.x * 16;
  const int n0 = blockIdx.y * 64;
  const int b  = blockIdx.z / 3;
  const int sel = blockIdx.z % 3;

  const __bf16* Wt  = (sel == 0) ? Wqt : (sel == 1) ? Wkt : Wvt;
  const float* bias = (sel == 0) ? bq : (sel == 1) ? bk : bv;

  const __bf16* arow = xbf + ((size_t)b * kT + (m0 + r)) * kC + 8 * hi;
  const __bf16* brow[4];
#pragma unroll
  for (int j = 0; j < 4; ++j)
    brow[j] = Wt + (size_t)(n0 + j * 16 + r) * kC + 8 * hi;

  v8f acc[4] = {v8f{}, v8f{}, v8f{}, v8f{}};

  for (int k0 = 0; k0 < kC; k0 += 32) {
    v16bf a = ldA(arow + k0);
#pragma unroll
    for (int j = 0; j < 4; ++j)
      acc[j] = wmma_bf16(a, ldB(brow[j] + k0), acc[j]);
  }

#pragma unroll
  for (int j = 0; j < 4; ++j) {
    const int co   = n0 + j * 16 + r;
    const int head = co >> 6;
    const int d    = co & 63;
    const float bv_ = bias[co];
    const size_t bh = (size_t)b * kNH + head;
#pragma unroll
    for (int i = 0; i < 8; ++i) {
      const int t = m0 + i + 8 * hi;
      const __bf16 v = (__bf16)(acc[j][i] + bv_);
      if (sel == 2)
        Vt[(bh * kHD + d) * kT + t] = v;          // (B,NH,HD,T)
      else if (sel == 0)
        Qb[(bh * kT + t) * kHD + d] = v;          // (B,NH,T,HD)
      else
        Kb[(bh * kT + t) * kHD + d] = v;
    }
  }
}

// ---------------------------------------------------------------------------
// Kernel 2: attention core for one (b, h, 16-query block).
// grid = (T/16, NH, B), block = 128 (4 waves).
// Pass 1: S = Q K^T / 8 -> LDS (bf16). Pass 2: row softmax (f32 stats),
// normalized P in place. Pass 3: Y = P V via WMMA (A from LDS, B from Vt).
// ---------------------------------------------------------------------------
__global__ void __launch_bounds__(128)
attn_kernel(const __bf16* __restrict__ Qb, const __bf16* __restrict__ Kb,
            const __bf16* __restrict__ Vt, __bf16* __restrict__ Ybf)
{
  __shared__ __bf16 Sb[16][kT];        // 32 KB
  __shared__ float  red[16][8];
  __shared__ float  rowmax[16];
  __shared__ float  rowsum[16];

  const int tid  = threadIdx.x;
  const int wave = tid >> 5;
  const int lane = tid & 31;
  const int r  = lane & 15;
  const int hi = lane >> 4;

  const int q0 = blockIdx.x * 16;
  const int h  = blockIdx.y;
  const int b  = blockIdx.z;
  const size_t bh = (size_t)b * kNH + h;
  const __bf16* Qp  = Qb + bh * kT * kHD;
  const __bf16* Kp  = Kb + bh * kT * kHD;
  const __bf16* Vtp = Vt + bh * kHD * kT;
  const float scale = 0.125f;          // 1/sqrt(64)

  // ---- Pass 1: each wave covers 256 key columns ----
  const __bf16* qrow = Qp + (size_t)(q0 + r) * kHD + 8 * hi;
  for (int kt = 0; kt < 16; ++kt) {
    const int n0 = wave * 256 + kt * 16;
    const __bf16* krow = Kp + (size_t)(n0 + r) * kHD + 8 * hi;
    v8f c = {};
#pragma unroll
    for (int kk = 0; kk < kHD; kk += 32)
      c = wmma_bf16(ldA(qrow + kk), ldB(krow + kk), c);
#pragma unroll
    for (int i = 0; i < 8; ++i)
      Sb[i + 8 * hi][n0 + r] = (__bf16)(c[i] * scale);
  }
  __syncthreads();

  // ---- Pass 2: softmax. 8 threads per row, 128 columns each. ----
  {
    const int row = tid >> 3;
    const int j   = tid & 7;
    const int c0  = j * 128;

    float m = -1e30f;
    for (int cc = 0; cc < 128; ++cc)
      m = fmaxf(m, (float)Sb[row][c0 + cc]);
    red[row][j] = m;
    __syncthreads();
    if (j == 0) {
      float mm = red[row][0];
      for (int q = 1; q < 8; ++q) mm = fmaxf(mm, red[row][q]);
      rowmax[row] = mm;
    }
    __syncthreads();

    const float mm = rowmax[row];
    float s = 0.f;
    for (int cc = 0; cc < 128; ++cc) {
      const float e = __expf((float)Sb[row][c0 + cc] - mm);
      Sb[row][c0 + cc] = (__bf16)e;
      s += e;
    }
    red[row][j] = s;
    __syncthreads();
    if (j == 0) {
      float ss = 0.f;
      for (int q = 0; q < 8; ++q) ss += red[row][q];
      rowsum[row] = ss;
    }
    __syncthreads();

    const float inv = 1.0f / rowsum[row];
    for (int cc = 0; cc < 128; ++cc)
      Sb[row][c0 + cc] = (__bf16)((float)Sb[row][c0 + cc] * inv);
  }
  __syncthreads();

  // ---- Pass 3: Y = P V. Wave w handles d-tile d0 = 16*w. ----
  {
    const int d0 = wave * 16;
    const __bf16* prow = &Sb[r][8 * hi];                       // LDS, b128 loads
    const __bf16* vrow = Vtp + (size_t)(d0 + r) * kT + 8 * hi;
    v8f c = {};
    for (int kk = 0; kk < kT; kk += 32)
      c = wmma_bf16(ldA(prow + kk), ldB(vrow + kk), c);
#pragma unroll
    for (int i = 0; i < 8; ++i) {
      const int t  = q0 + i + 8 * hi;
      const int cc = h * kHD + d0 + r;
      Ybf[((size_t)b * kT + t) * kC + cc] = (__bf16)c[i];
    }
  }
}

// ---------------------------------------------------------------------------
// Kernel 3: output projection + bias + residual, store to (B,C,H,W) f32.
// grid = (T/16, C/64, B), block = 32 (one wave).
// ---------------------------------------------------------------------------
__global__ void __launch_bounds__(32)
out_proj_kernel(const __bf16* __restrict__ Ybf, const __bf16* __restrict__ Wot,
                const float* __restrict__ bo, const float* __restrict__ x,
                float* __restrict__ out)
{
  const int lane = threadIdx.x & 31;
  const int r  = lane & 15;
  const int hi = lane >> 4;
  const int m0 = blockIdx.x * 16;
  const int n0 = blockIdx.y * 64;
  const int b  = blockIdx.z;

  const __bf16* arow = Ybf + ((size_t)b * kT + (m0 + r)) * kC + 8 * hi;
  const __bf16* brow[4];
#pragma unroll
  for (int j = 0; j < 4; ++j)
    brow[j] = Wot + (size_t)(n0 + j * 16 + r) * kC + 8 * hi;

  v8f acc[4] = {v8f{}, v8f{}, v8f{}, v8f{}};

  for (int k0 = 0; k0 < kC; k0 += 32) {
    v16bf a = ldA(arow + k0);
#pragma unroll
    for (int j = 0; j < 4; ++j)
      acc[j] = wmma_bf16(a, ldB(brow[j] + k0), acc[j]);
  }

#pragma unroll
  for (int j = 0; j < 4; ++j) {
    const int co = n0 + j * 16 + r;
    const float bv_ = bo[co];
#pragma unroll
    for (int i = 0; i < 8; ++i) {
      const int t = m0 + i + 8 * hi;
      const size_t idx = ((size_t)b * kC + co) * kT + t;
      out[idx] = x[idx] + acc[j][i] + bv_;
    }
  }
}

// ---------------------------------------------------------------------------
// Launcher
// ---------------------------------------------------------------------------
extern "C" void kernel_launch(void* const* d_in, const int* in_sizes, int n_in,
                              void* d_out, int out_size, void* d_ws, size_t ws_size,
                              hipStream_t stream) {
  const float* x  = (const float*)d_in[0];
  const float* Wq = (const float*)d_in[1];
  const float* bq = (const float*)d_in[2];
  const float* Wk = (const float*)d_in[3];
  const float* bk = (const float*)d_in[4];
  const float* Wv = (const float*)d_in[5];
  const float* bv = (const float*)d_in[6];
  const float* Wo = (const float*)d_in[7];
  const float* bo = (const float*)d_in[8];
  float* out = (float*)d_out;

  const size_t nTok = (size_t)kB * kT * kC;   // 4,194,304 elems
  const size_t nW   = (size_t)kC * kC;        //   262,144 elems

  __bf16* ws  = (__bf16*)d_ws;
  __bf16* xbf = ws;                 // (B,T,C)
  __bf16* Wqt = xbf + nTok;         // (Cout,Cin) x4
  __bf16* Wkt = Wqt + nW;
  __bf16* Wvt = Wkt + nW;
  __bf16* Wot = Wvt + nW;
  __bf16* Qb  = Wot + nW;           // (B,NH,T,HD)
  __bf16* Kb  = Qb + nTok;          // (B,NH,T,HD)
  __bf16* Vt  = Kb + nTok;          // (B,NH,HD,T)
  __bf16* Ybf = Vt + nTok;          // (B,T,C)

  // Pre-pass: bf16 conversion (+ transposes folded in).
  xpose_x_kernel<<<dim3(kT / 32, kC / 32, kB), 256, 0, stream>>>(x, xbf);
  xpose_w_kernel<<<dim3(kC / 32, kC / 32, 4), 256, 0, stream>>>(
      Wq, Wk, Wv, Wo, Wqt, Wkt, Wvt, Wot);

  qkv_proj_kernel<<<dim3(kT / 16, kC / 64, kB * 3), 32, 0, stream>>>(
      xbf, Wqt, bq, Wkt, bk, Wvt, bv, Qb, Kb, Vt);

  attn_kernel<<<dim3(kT / 16, kNH, kB), 128, 0, stream>>>(Qb, Kb, Vt, Ybf);

  out_proj_kernel<<<dim3(kT / 16, kC / 64, kB), 32, 0, stream>>>(
      Ybf, Wot, bo, x, out);
}
